// MultiHeadLatentAttention_65352222375939
// MI455X (gfx1250) — compile-verified, matrix operand order checked by reading, not run on hardware
//
#include <hip/hip_runtime.h>
#include <hip/hip_bf16.h>

// ---------------------------------------------------------------------------
// CDNA5 (gfx1250) MLA: bf16 WMMA GEMMs (async-LDS staged) + flash attention.
// ---------------------------------------------------------------------------

typedef __attribute__((ext_vector_type(16))) __bf16 v16bf;
typedef __attribute__((ext_vector_type(8)))  __bf16 v8bf;
typedef __attribute__((ext_vector_type(8)))  float  v8f;

static __device__ inline v8bf ld8(const __bf16* p) {
    return *reinterpret_cast<const v8bf*>(p);
}

// Build a 16x32 (A) or 32x16-as-row (B) bf16 fragment from a 32-element row.
// Per ISA 7.12.2: lane half selects K offset {0,8}; VGPRs 0-3 cover K base+0..7,
// VGPRs 4-7 cover K base+16..23 -> two contiguous 16B loads per lane.
static __device__ inline v16bf frag32(const __bf16* rowp, int hofs) {
    v8bf lo = ld8(rowp + hofs);
    v8bf hi = ld8(rowp + 16 + hofs);
    return __builtin_shufflevector(lo, hi, 0,1,2,3,4,5,6,7,8,9,10,11,12,13,14,15);
}

static __device__ inline v8f wmma_bf16(v16bf a, v16bf b, v8f c) {
    return __builtin_amdgcn_wmma_f32_16x16x32_bf16(
        /*neg_a=*/false, a, /*neg_b=*/false, b,
        /*c_mod=*/(short)0, c, /*reuse_a=*/false, /*reuse_b=*/false);
}

// Async global->LDS 16B copy (ASYNCcnt-tracked, CDNA5 §10.7/§15.18).
// lds_addr: LDS byte address (low 32 bits of flat shared pointer; the LDS
// aperture maps flat->LDS by truncating to addr[31:0]).
static __device__ inline void async_g2l_b128(unsigned int lds_addr, const __bf16* gp) {
    asm volatile("global_load_async_to_lds_b128 %0, %1, off"
                 :: "v"(lds_addr), "v"((unsigned long long)(uintptr_t)gp)
                 : "memory");
}
#define WAIT_ASYNC()    asm volatile("s_wait_asynccnt 0x0" ::: "memory")
#define WAVE_DS_FENCE() asm volatile("s_wait_dscnt 0x0"   ::: "memory")

// ---------------------------------------------------------------------------
// f32 -> bf16 conversion (grid-stride)
// ---------------------------------------------------------------------------
__global__ void cvt_f32_bf16(const float* __restrict__ s, __bf16* __restrict__ d, size_t n) {
    size_t i  = (size_t)blockIdx.x * blockDim.x + threadIdx.x;
    size_t st = (size_t)gridDim.x * blockDim.x;
    for (; i < n; i += st) d[i] = (__bf16)s[i];
}

// ---------------------------------------------------------------------------
// Row-major bf16 GEMM: C(MxN,f32) = A(MxK) @ B(KxN).
// 8 waves/block, 128x64 macro tile, wave tile 32x32 (4 WMMA / 32-k step).
// A tile staged via GLOBAL_LOAD_ASYNC_TO_LDS_B128; B tile transposed in LDS.
// M % 128 == 0, N % 64 == 0, K % 32 == 0 (true for all call sites here).
// ---------------------------------------------------------------------------
__global__ void __launch_bounds__(256)
gemm_bf16(const __bf16* __restrict__ A, const __bf16* __restrict__ B,
          float* __restrict__ C, int M, int N, int K) {
    __shared__ __bf16 lA[128][32];  // A tile, row-major
    __shared__ __bf16 lBt[64][32];  // B tile transposed: lBt[n][k]

    const int tid  = threadIdx.x;
    const int lane = tid & 31;
    const int wave = tid >> 5;
    const int wm   = wave >> 1;      // 0..3 -> 32-row slab
    const int wn   = wave & 1;       // 0..1 -> 32-col slab
    const int m0   = blockIdx.y * 128;
    const int n0   = blockIdx.x * 64;
    const int half = lane >> 4;
    const int hofs = half * 8;
    const int mr   = lane & 15;

    // A coop staging: thread -> (row 0..127, 16 elems); two async b128 copies
    const int arow = tid >> 1, acol = (tid & 1) * 16;
    const unsigned int ldsA0 = (unsigned int)(uintptr_t)&lA[arow][acol];
    // B coop staging: thread -> (k row 0..31, 8 n-cols), transposed scatter
    const int bkk = tid >> 3, bn8 = (tid & 7) * 8;

    v8f c00 = {}, c01 = {}, c10 = {}, c11 = {};

    for (int k0 = 0; k0 < K; k0 += 32) {
        const __bf16* ga = A + (size_t)(m0 + arow) * K + k0 + acol;
        async_g2l_b128(ldsA0,      ga);
        async_g2l_b128(ldsA0 + 16, ga + 8);

        v8bf bv = ld8(B + (size_t)(k0 + bkk) * N + n0 + bn8);
        #pragma unroll
        for (int i = 0; i < 8; ++i) lBt[bn8 + i][bkk] = bv[i];

        if (k0 + 32 < K) {  // warm L2->WGP path for the next slab
            __builtin_prefetch(A + (size_t)(m0 + arow) * K + k0 + 32 + acol, 0, 1);
            __builtin_prefetch(B + (size_t)(k0 + 32 + bkk) * N + n0 + bn8, 0, 1);
        }

        WAIT_ASYNC();
        __syncthreads();

        v16bf a0 = frag32(&lA[wm * 32 + mr][0],      hofs);
        v16bf a1 = frag32(&lA[wm * 32 + 16 + mr][0], hofs);
        v16bf b0 = frag32(&lBt[wn * 32 + mr][0],     hofs);
        v16bf b1 = frag32(&lBt[wn * 32 + 16 + mr][0], hofs);
        c00 = wmma_bf16(a0, b0, c00);
        c01 = wmma_bf16(a0, b1, c01);
        c10 = wmma_bf16(a1, b0, c10);
        c11 = wmma_bf16(a1, b1, c11);
        __syncthreads();
    }

    // D layout: lane holds column n=lane%16, VGPR r holds row r + 8*(lane>=16)
    #pragma unroll
    for (int r = 0; r < 8; ++r) {
        const size_t row0 = (size_t)(m0 + wm * 32 + r + 8 * half);
        const size_t row1 = row0 + 16;
        C[row0 * N + n0 + wn * 32 + mr]      = c00[r];
        C[row0 * N + n0 + wn * 32 + 16 + mr] = c01[r];
        C[row1 * N + n0 + wn * 32 + mr]      = c10[r];
        C[row1 * N + n0 + wn * 32 + 16 + mr] = c11[r];
    }
}

// ---------------------------------------------------------------------------
// RoPE + concat: q_c/q_r/k_c/k_r (B,S,nh,64) f32 -> Q,K (B,nh,S,128) bf16.
// Folds attention scale 1/sqrt(128) into Q.
// ---------------------------------------------------------------------------
__global__ void rope_concat_qk(const float* __restrict__ q_c, const float* __restrict__ q_r,
                               const float* __restrict__ k_c, const float* __restrict__ k_r,
                               __bf16* __restrict__ Qo, __bf16* __restrict__ Ko,
                               int Bn, int S, int nh) {
    const float scale = 0.08838834764831845f;  // 128^-0.5
    size_t i  = (size_t)blockIdx.x * blockDim.x + threadIdx.x;
    size_t st = (size_t)gridDim.x * blockDim.x;
    const size_t total = (size_t)Bn * nh * S * 128;
    for (; i < total; i += st) {
        int d = (int)(i & 127);
        size_t rest = i >> 7;
        int s = (int)(rest % S); rest /= S;
        int h = (int)(rest % nh);
        int b = (int)(rest / nh);
        size_t inBase = (((size_t)b * S + s) * nh + h) * 64;
        float qv, kv;
        if (d < 64) {
            qv = q_c[inBase + d];
            kv = k_c[inBase + d];
        } else {
            int dr = d - 64;
            int fi = dr & 31;
            float inv = __powf(10000.0f, -(float)(2 * fi) * (1.0f / 64.0f));
            float ang = (float)s * inv;
            float c = __cosf(ang), sn = __sinf(ang);
            float qx = q_r[inBase + dr];
            float kx = k_r[inBase + dr];
            float qrot = (dr < 32) ? -q_r[inBase + dr + 32] : q_r[inBase + dr - 32];
            float krot = (dr < 32) ? -k_r[inBase + dr + 32] : k_r[inBase + dr - 32];
            qv = qx * c + qrot * sn;
            kv = kx * c + krot * sn;
        }
        size_t o = (((size_t)b * nh + h) * S + s) * 128 + d;
        Qo[o] = (__bf16)(qv * scale);
        Ko[o] = (__bf16)kv;
    }
}

// V (B,S,nh,128) f32 -> Vt (B,nh,128,S) bf16: PV B-fragments become contiguous rows.
__global__ void transpose_v(const float* __restrict__ V, __bf16* __restrict__ Vt,
                            int Bn, int S, int nh) {
    size_t i  = (size_t)blockIdx.x * blockDim.x + threadIdx.x;
    size_t st = (size_t)gridDim.x * blockDim.x;
    const size_t total = (size_t)Bn * nh * 128 * S;
    for (; i < total; i += st) {
        int s = (int)(i % S);
        size_t rest = i / S;
        int d = (int)(rest & 127); rest >>= 7;
        int h = (int)(rest % nh);
        int b = (int)(rest / nh);
        Vt[i] = (__bf16)V[(((size_t)b * S + s) * nh + h) * 128 + d];
    }
}

// Y (B,nh,S,128) f32 -> Yt (B,S,H) bf16
__global__ void transpose_y(const float* __restrict__ Y, __bf16* __restrict__ Yt,
                            int Bn, int S, int nh) {
    size_t i  = (size_t)blockIdx.x * blockDim.x + threadIdx.x;
    size_t st = (size_t)gridDim.x * blockDim.x;
    const int H = nh * 128;
    const size_t total = (size_t)Bn * S * H;
    for (; i < total; i += st) {
        int hd = (int)(i % H);
        int d = hd & 127, h = hd >> 7;
        size_t rest = i / H;
        int s = (int)(rest % S);
        int b = (int)(rest / S);
        Yt[i] = (__bf16)Y[(((size_t)b * nh + h) * S + s) * 128 + d];
    }
}

// ---------------------------------------------------------------------------
// Flash attention (causal). Q,K: (B*nh, S, 128) bf16 (Q pre-scaled),
// Vt: (B*nh, 128, S) bf16. Y: (B*nh, S, 128) f32.
// One wave per 16-row q-tile; 32 keys/step: 8 score WMMAs + 8 PV WMMAs.
// ---------------------------------------------------------------------------
__global__ void __launch_bounds__(128)
flash_attn(const __bf16* __restrict__ Q, const __bf16* __restrict__ Kmat,
           const __bf16* __restrict__ Vt, float* __restrict__ Y, int S) {
    constexpr int WPB = 4;
    __shared__ __bf16 pT[WPB][16][32];   // per-wave P transpose bounce

    const int lane = threadIdx.x & 31;
    const int wave = threadIdx.x >> 5;
    const int bh   = blockIdx.y;
    const int q0   = (blockIdx.x * WPB + wave) * 16;
    const int half = lane >> 4;
    const int hofs = half * 8;
    const int ln   = lane & 15;

    const __bf16* Qb = Q    + (size_t)bh * S * 128;
    const __bf16* Kb = Kmat + (size_t)bh * S * 128;
    const __bf16* Vb = Vt   + (size_t)bh * 128 * S;

    // Q A-fragments: K=128 -> 4 chained 16x32 fragments
    v16bf qf[4];
    #pragma unroll
    for (int j = 0; j < 4; ++j)
        qf[j] = frag32(Qb + (size_t)(q0 + ln) * 128 + j * 32, hofs);

    v8f acc[8] = {};
    float mrow[8], lrow[8];
    #pragma unroll
    for (int r = 0; r < 8; ++r) { mrow[r] = -3.0e38f; lrow[r] = 0.0f; }

    for (int kb = 0; kb <= q0 + 15; kb += 32) {
        // scores for 32 keys: two 16x16 D tiles, K=128 chains
        v8f s0 = {}, s1 = {};
        #pragma unroll
        for (int j = 0; j < 4; ++j) {
            v16bf k0f = frag32(Kb + (size_t)(kb + ln) * 128 + j * 32, hofs);
            v16bf k1f = frag32(Kb + (size_t)(kb + 16 + ln) * 128 + j * 32, hofs);
            s0 = wmma_bf16(qf[j], k0f, s0);
            s1 = wmma_bf16(qf[j], k1f, s1);
        }

        // causal mask: lane column = key, VGPR r = query row r + 8*half
        #pragma unroll
        for (int r = 0; r < 8; ++r) {
            const int qrow = q0 + 8 * half + r;
            if (kb + ln      > qrow) s0[r] = -3.0e38f;
            if (kb + 16 + ln > qrow) s1[r] = -3.0e38f;
        }

        // online softmax: row stats reduced across the 16 lanes of each half
        float p0a[8], p1a[8], alpha[8];
        #pragma unroll
        for (int r = 0; r < 8; ++r) {
            float rm = fmaxf(s0[r], s1[r]);
            for (int m = 8; m >= 1; m >>= 1) rm = fmaxf(rm, __shfl_xor(rm, m, 16));
            float mnew = fmaxf(mrow[r], rm);
            float a  = __expf(mrow[r] - mnew);
            float p0 = __expf(s0[r] - mnew);
            float p1 = __expf(s1[r] - mnew);
            float rs = p0 + p1;
            for (int m = 8; m >= 1; m >>= 1) rs += __shfl_xor(rs, m, 16);
            lrow[r] = lrow[r] * a + rs;
            mrow[r] = mnew;
            alpha[r] = a; p0a[r] = p0; p1a[r] = p1;
        }
        #pragma unroll
        for (int t = 0; t < 8; ++t)
            #pragma unroll
            for (int r = 0; r < 8; ++r) acc[t][r] *= alpha[r];

        // transpose P (D layout -> A layout) through wave-private LDS
        #pragma unroll
        for (int r = 0; r < 8; ++r) {
            pT[wave][8 * half + r][ln]      = (__bf16)p0a[r];
            pT[wave][8 * half + r][16 + ln] = (__bf16)p1a[r];
        }
        WAVE_DS_FENCE();                       // cross-lane LDS visibility (wave-sync)
        v16bf pf = frag32(&pT[wave][ln][0], hofs);

        // PV: B fragments are contiguous rows of Vt
        #pragma unroll
        for (int t = 0; t < 8; ++t) {
            v16bf vf = frag32(Vb + (size_t)(t * 16 + ln) * S + kb, hofs);
            acc[t] = wmma_bf16(pf, vf, acc[t]);
        }
        WAVE_DS_FENCE();
    }

    #pragma unroll
    for (int t = 0; t < 8; ++t)
        #pragma unroll
        for (int r = 0; r < 8; ++r) {
            const size_t row = (size_t)(q0 + 8 * half + r);
            Y[(size_t)bh * S * 128 + row * 128 + t * 16 + ln] = acc[t][r] / lrow[r];
        }
}

// ---------------------------------------------------------------------------
// Host orchestration
// ---------------------------------------------------------------------------
extern "C" void kernel_launch(void* const* d_in, const int* in_sizes, int n_in,
                              void* d_out, int out_size, void* d_ws, size_t ws_size,
                              hipStream_t stream) {
    (void)in_sizes; (void)n_in; (void)out_size; (void)ws_size;
    const int Bn = 2, S = 2048, H = 2048, nh = 16;
    const int BS = Bn * S;           // 4096
    const int L = H / 8;             // 256
    const int Hh = H / 2;            // 1024

    const float* hs  = (const float*)d_in[0];
    const float* wkv = (const float*)d_in[1];
    const float* wqd = (const float*)d_in[2];
    const float* wku = (const float*)d_in[3];
    const float* wqu = (const float*)d_in[4];
    const float* wvu = (const float*)d_in[5];
    const float* wrk = (const float*)d_in[6];
    const float* wrq = (const float*)d_in[7];
    const float* wo  = (const float*)d_in[8];

    size_t off = 0;
    auto alloc = [&](size_t bytes) -> void* {
        void* p = (char*)d_ws + off;
        off += (bytes + 255) & ~(size_t)255;
        return p;
    };
    auto nBF = [&](size_t n) { return (__bf16*)alloc(n * 2); };
    auto nF  = [&](size_t n) { return (float*)alloc(n * 4); };

    __bf16* hs_bf  = nBF((size_t)BS * H);
    __bf16* wkv_bf = nBF((size_t)H * L);
    __bf16* wqd_bf = nBF((size_t)H * L);
    __bf16* wku_bf = nBF((size_t)L * Hh);
    __bf16* wqu_bf = nBF((size_t)L * Hh);
    __bf16* wvu_bf = nBF((size_t)L * H);
    __bf16* wrk_bf = nBF((size_t)H * Hh);
    __bf16* wrq_bf = nBF((size_t)L * Hh);
    __bf16* wo_bf  = nBF((size_t)H * H);
    float*  kvd_f  = nF((size_t)BS * L);
    __bf16* kvd_bf = nBF((size_t)BS * L);
    float*  qd_f   = nF((size_t)BS * L);
    __bf16* qd_bf  = nBF((size_t)BS * L);
    float*  kc_f   = nF((size_t)BS * Hh);
    float*  qc_f   = nF((size_t)BS * Hh);
    float*  v_f    = nF((size_t)BS * H);
    float*  kr_f   = nF((size_t)BS * Hh);
    float*  qr_f   = nF((size_t)BS * Hh);
    __bf16* q_all  = nBF((size_t)Bn * nh * S * 128);
    __bf16* k_all  = nBF((size_t)Bn * nh * S * 128);
    __bf16* vT     = nBF((size_t)Bn * nh * 128 * S);
    float*  y_f    = nF((size_t)Bn * nh * S * 128);
    __bf16* yt_bf  = nBF((size_t)BS * H);

    const int CT = 256, CG = 2048;
    cvt_f32_bf16<<<CG, CT, 0, stream>>>(hs,  hs_bf,  (size_t)BS * H);
    cvt_f32_bf16<<<CG, CT, 0, stream>>>(wkv, wkv_bf, (size_t)H * L);
    cvt_f32_bf16<<<CG, CT, 0, stream>>>(wqd, wqd_bf, (size_t)H * L);
    cvt_f32_bf16<<<CG, CT, 0, stream>>>(wku, wku_bf, (size_t)L * Hh);
    cvt_f32_bf16<<<CG, CT, 0, stream>>>(wqu, wqu_bf, (size_t)L * Hh);
    cvt_f32_bf16<<<CG, CT, 0, stream>>>(wvu, wvu_bf, (size_t)L * H);
    cvt_f32_bf16<<<CG, CT, 0, stream>>>(wrk, wrk_bf, (size_t)H * Hh);
    cvt_f32_bf16<<<CG, CT, 0, stream>>>(wrq, wrq_bf, (size_t)L * Hh);
    cvt_f32_bf16<<<CG, CT, 0, stream>>>(wo,  wo_bf,  (size_t)H * H);

    // down-projections
    gemm_bf16<<<dim3(L / 64, BS / 128), 256, 0, stream>>>(hs_bf, wkv_bf, kvd_f, BS, L, H);
    gemm_bf16<<<dim3(L / 64, BS / 128), 256, 0, stream>>>(hs_bf, wqd_bf, qd_f,  BS, L, H);
    cvt_f32_bf16<<<CG, CT, 0, stream>>>(kvd_f, kvd_bf, (size_t)BS * L);
    cvt_f32_bf16<<<CG, CT, 0, stream>>>(qd_f,  qd_bf,  (size_t)BS * L);
    // up-projections + rope branches
    gemm_bf16<<<dim3(Hh / 64, BS / 128), 256, 0, stream>>>(kvd_bf, wku_bf, kc_f, BS, Hh, L);
    gemm_bf16<<<dim3(Hh / 64, BS / 128), 256, 0, stream>>>(qd_bf,  wqu_bf, qc_f, BS, Hh, L);
    gemm_bf16<<<dim3(H  / 64, BS / 128), 256, 0, stream>>>(kvd_bf, wvu_bf, v_f,  BS, H,  L);
    gemm_bf16<<<dim3(Hh / 64, BS / 128), 256, 0, stream>>>(hs_bf,  wrk_bf, kr_f, BS, Hh, H);
    gemm_bf16<<<dim3(Hh / 64, BS / 128), 256, 0, stream>>>(qd_bf,  wrq_bf, qr_f, BS, Hh, L);
    // rope + layout
    rope_concat_qk<<<CG, CT, 0, stream>>>(qc_f, qr_f, kc_f, kr_f, q_all, k_all, Bn, S, nh);
    transpose_v<<<CG, CT, 0, stream>>>(v_f, vT, Bn, S, nh);
    // attention
    flash_attn<<<dim3(S / 64, Bn * nh), 128, 0, stream>>>(q_all, k_all, vT, y_f, S);
    // output projection
    transpose_y<<<CG, CT, 0, stream>>>(y_f, yt_bf, Bn, S, nh);
    gemm_bf16<<<dim3(H / 64, BS / 128), 256, 0, stream>>>(yt_bf, wo_bf, (float*)d_out, BS, H, H);
}
